// AttentionHead_9045201125978
// MI455X (gfx1250) — matrix operand order repaired and checked by reading, not compiled, not run on hardware
//
#include <hip/hip_runtime.h>

typedef __attribute__((ext_vector_type(16))) _Float16 v16h;
typedef __attribute__((ext_vector_type(8)))  _Float16 v8h;
typedef __attribute__((ext_vector_type(8)))  float    v8f;
typedef __attribute__((ext_vector_type(4)))  unsigned int v4u;
typedef __attribute__((ext_vector_type(8)))  int      v8i;
typedef __attribute__((ext_vector_type(4)))  int      v4i;

#define HEADSZ 128
#define EMB    1024
#define SEQ    2048
#define NBATCH 8

#ifndef __has_builtin
#define __has_builtin(x) 0
#endif
#if __has_builtin(__builtin_amdgcn_tensor_load_to_lds) && \
    __has_builtin(__builtin_amdgcn_s_wait_tensorcnt)
#define USE_TDM 1
#else
#define USE_TDM 0
#endif

__device__ __forceinline__ v8f wmma16(v16h a, v16h b, v8f c) {
    // D = A(16x32 f16) * B(32x16 f16) + C(16x16 f32)
    return __builtin_amdgcn_wmma_f32_16x16x32_f16(
        false, a, false, b, (short)0, c, false, false);
}

#if USE_TDM
// Build a 2-D TDM descriptor (D#) per CDNA5 ISA §8 and issue TENSOR_LOAD_TO_LDS.
// tile_d0 = row length (elements), tile_d1 = #rows, stride0 = global row stride.
__device__ __forceinline__ void tdm_load_2d(unsigned lds_byte_addr, const void* gp,
                                            unsigned tile_d0, unsigned tile_d1,
                                            unsigned tensor_d0, unsigned tensor_d1,
                                            unsigned long long stride0)
{
    unsigned long long ga = (unsigned long long)(size_t)gp;
    v4u g0;
    g0[0] = 1u;                                            // count=1, user mode
    g0[1] = lds_byte_addr;                                 // lds_addr (bytes)
    g0[2] = (unsigned)(ga & 0xFFFFFFFFu);                  // global_addr[31:0]
    g0[3] = (unsigned)((ga >> 32) & 0x01FFFFFFu) | (2u << 30);  // ga[56:32], type=2
    v8i g1;
    g1[0] = (int)(1u << 16);                               // data_size=1 (2 bytes)
    g1[1] = (int)((tensor_d0 & 0xFFFFu) << 16);            // tensor_dim0[15:0]
    g1[2] = (int)((tensor_d0 >> 16) | ((tensor_d1 & 0xFFFFu) << 16));
    g1[3] = (int)((tensor_d1 >> 16) | (tile_d0 << 16));    // tile_dim0
    g1[4] = (int)(tile_d1 & 0xFFFFu);                      // tile_dim1 (tile_dim2=0)
    g1[5] = (int)(stride0 & 0xFFFFFFFFu);                  // tensor_dim0_stride lo
    g1[6] = (int)((stride0 >> 32) & 0xFFFFu);              // stride hi (dim1_stride=0)
    g1[7] = 0;
    v4i g2 = {};                                           // 2-D: dims 2/3 unused
    v4i g3 = {};
#if __clang_major__ >= 23
    v8i g4 = {};
    __builtin_amdgcn_tensor_load_to_lds(g0, g1, g2, g3, g4, 0);
#else
    __builtin_amdgcn_tensor_load_to_lds(g0, g1, g2, g3, 0);
#endif
}
#endif

// ---------------------------------------------------------------------------
// Prep: W_key / W_value (f32 [1024][128]) -> f16 transposed [128][1024]
// ---------------------------------------------------------------------------
__global__ __launch_bounds__(256)
void ah_cvt_w_kernel(const float* __restrict__ Wk, const float* __restrict__ Wv,
                     _Float16* __restrict__ WkT, _Float16* __restrict__ WvT)
{
    const int idx = blockIdx.x * 256 + threadIdx.x;   // 0 .. 131071
    const int k = idx >> 7;
    const int c = idx & 127;
    WkT[(size_t)c * EMB + k] = (_Float16)Wk[idx];
    WvT[(size_t)c * EMB + k] = (_Float16)Wv[idx];
}

// ---------------------------------------------------------------------------
// Kernel 1: K = x@W_key, V = x@W_value; x strip staged once in LDS as f16.
//   K row-major [B*T][128] f16; V transposed [B][128][T] f16.
// grid=(BT/16), block=256 (8 waves; wave w -> head-col tile w)
// ---------------------------------------------------------------------------
__global__ __launch_bounds__(256)
void ah_proj_kernel(const float* __restrict__ x,
                    const _Float16* __restrict__ WkT,
                    const _Float16* __restrict__ WvT,
                    _Float16* __restrict__ Kf,
                    _Float16* __restrict__ Vt)
{
    __shared__ __align__(16) _Float16 xs[16 * EMB];   // 32 KB
    const int tid = threadIdx.x;
    const int r0  = blockIdx.x * 16;

    for (int i = tid * 8; i < 16 * EMB; i += 256 * 8) {
        const float* p = x + (size_t)r0 * EMB + i;
        float4 a = *(const float4*)p;
        float4 b = *(const float4*)(p + 4);
        v8h h;
        h[0]=(_Float16)a.x; h[1]=(_Float16)a.y; h[2]=(_Float16)a.z; h[3]=(_Float16)a.w;
        h[4]=(_Float16)b.x; h[5]=(_Float16)b.y; h[6]=(_Float16)b.z; h[7]=(_Float16)b.w;
        *(v8h*)(xs + i) = h;
    }
    __syncthreads();

    const int lane = tid & 31;
    const int wid  = tid >> 5;
    const int m    = lane & 15;
    const int hh   = lane >> 4;
    const int col  = wid * 16 + m;

    v8f accK = {}, accV = {};
    const _Float16* xr  = xs + m * EMB;
    const _Float16* wkp = WkT + (size_t)col * EMB;
    const _Float16* wvp = WvT + (size_t)col * EMB;

    for (int k0 = 0; k0 < EMB; k0 += 32) {
        v16h A;
        v8h lo = *(const v8h*)(xr + k0 + hh * 16);
        v8h hi = *(const v8h*)(xr + k0 + 8 + hh * 16);
        #pragma unroll
        for (int i = 0; i < 8; ++i) { A[i] = lo[i]; A[8 + i] = hi[i]; }
        v16h bk = *(const v16h*)(wkp + k0 + 16 * hh);
        v16h bv = *(const v16h*)(wvp + k0 + 16 * hh);
        accK = wmma16(A, bk, accK);
        accV = wmma16(A, bv, accV);
    }

    _Float16* ok = Kf + (size_t)r0 * HEADSZ;
    #pragma unroll
    for (int r = 0; r < 8; ++r)
        ok[(size_t)(r + 8 * hh) * HEADSZ + col] = (_Float16)accK[r];

    const int b  = r0 / SEQ;
    const int t0 = r0 % SEQ;
    v8h pk;
    #pragma unroll
    for (int r = 0; r < 8; ++r) pk[r] = (_Float16)accV[r];
    _Float16* ov = Vt + (size_t)b * HEADSZ * SEQ + (size_t)col * SEQ + t0 + 8 * hh;
    *(v8h*)ov = pk;
}

// ---------------------------------------------------------------------------
// Kernel 2: fused causal flash attention.
// Block = 64-query tile (4 waves x 16 rows) marching over shared 64-key blocks.
// K/V tiles staged in LDS (TDM async double-buffer when available).
// grid=(B, SEQ/64) with LPT ordering, block=128.
// ---------------------------------------------------------------------------
__global__ __launch_bounds__(128)
void ah_attn_kernel(const _Float16* __restrict__ Kf,
                    const _Float16* __restrict__ Vt,
                    float* __restrict__ out)
{
    __shared__ __align__(32) _Float16 Kst[2][64 * HEADSZ];   // 32 KB  [key][h]
    __shared__ __align__(32) _Float16 Vst[2][HEADSZ * 64];   // 32 KB  [h][key]
    __shared__ __align__(16) _Float16 ldsP[4][16 * 64];      //  8 KB

    const int tid  = threadIdx.x;
    const int lane = tid & 31;
    const int wid  = tid >> 5;
    const int m    = lane & 15;
    const int hh   = lane >> 4;
    const int b    = blockIdx.x;
    // LPT: longest (highest-q) tiles launch first
    const int q0blk = 64 * ((int)gridDim.y - 1 - (int)blockIdx.y);
    const int q0    = q0blk + 16 * wid;

    const _Float16* Kb = Kf + (size_t)b * SEQ * HEADSZ;
    const _Float16* Vb = Vt + (size_t)b * HEADSZ * SEQ;
    _Float16* myP = ldsP[wid];

    // Q A-frags (Q == K per reference bug), 4 h-slices of 32
    v16h aq[4];
    const _Float16* qrow = Kb + (size_t)(q0 + m) * HEADSZ;
    #pragma unroll
    for (int s = 0; s < 4; ++s) {
        v8h lo = *(const v8h*)(qrow + 32 * s + hh * 16);
        v8h hi = *(const v8h*)(qrow + 32 * s + 8 + hh * 16);
        #pragma unroll
        for (int i = 0; i < 8; ++i) { aq[s][i] = lo[i]; aq[s][8 + i] = hi[i]; }
    }

    float mrow[8], lrow[8];
    #pragma unroll
    for (int r = 0; r < 8; ++r) { mrow[r] = -3.0e30f; lrow[r] = 0.0f; }
    v8f O[8] = {};

    const float scale = 0.08838834764831845f;   // 1/sqrt(128)
    const int nkb = q0blk / 64 + 1;

    // ---- stage first K/V tile ----
#if USE_TDM
    if (wid == 0) {
        tdm_load_2d((unsigned)(size_t)&Kst[0][0], Kb, HEADSZ, 64, HEADSZ, SEQ, HEADSZ);
        tdm_load_2d((unsigned)(size_t)&Vst[0][0], Vb, 64, HEADSZ, SEQ, HEADSZ, SEQ);
        __builtin_amdgcn_s_wait_tensorcnt(0);
    }
    __syncthreads();
#else
    for (int i = tid; i < 1024; i += 128)
        *(v8h*)(&Kst[0][0] + i * 8) = *(const v8h*)(Kb + (size_t)i * 8);
    for (int i = tid; i < 1024; i += 128) {
        int h = i >> 3, c = i & 7;
        *(v8h*)(&Vst[0][0] + h * 64 + c * 8) = *(const v8h*)(Vb + (size_t)h * SEQ + c * 8);
    }
    __syncthreads();
#endif

    int buf = 0;
    for (int ib = 0; ib < nkb; ++ib) {
        const int kb = ib * 64;
        const _Float16* Ks = Kst[buf];
        const _Float16* Vs = Vst[buf];

        // async-prefetch next tile into the other buffer
        if (ib + 1 < nkb) {
            const int kn = kb + 64;
#if USE_TDM
            if (wid == 0) {
                tdm_load_2d((unsigned)(size_t)&Kst[buf ^ 1][0],
                            Kb + (size_t)kn * HEADSZ, HEADSZ, 64, HEADSZ, SEQ, HEADSZ);
                tdm_load_2d((unsigned)(size_t)&Vst[buf ^ 1][0],
                            Vb + kn, 64, HEADSZ, SEQ, HEADSZ, SEQ);
            }
#else
            for (int i = tid; i < 1024; i += 128)
                *(v8h*)(&Kst[buf ^ 1][0] + i * 8) =
                    *(const v8h*)(Kb + (size_t)(kn) * HEADSZ + i * 8);
            for (int i = tid; i < 1024; i += 128) {
                int h = i >> 3, c = i & 7;
                *(v8h*)(&Vst[buf ^ 1][0] + h * 64 + c * 8) =
                    *(const v8h*)(Vb + (size_t)h * SEQ + kn + c * 8);
            }
#endif
        }

        // ---- S = Q @ K^T for 64 keys (four 16-col D tiles), K from LDS ----
        v8f S[4] = {};
        #pragma unroll
        for (int s = 0; s < 4; ++s) {
            #pragma unroll
            for (int n = 0; n < 4; ++n) {
                const _Float16* pk = Ks + (size_t)(16 * n + m) * HEADSZ + 32 * s + 16 * hh;
                S[n] = wmma16(aq[s], *(const v16h*)pk, S[n]);
            }
        }

        // ---- scale, causal mask, online softmax; write P to LDS ----
        float alpha[8];
        #pragma unroll
        for (int r = 0; r < 8; ++r) {
            const int q = q0 + r + 8 * hh;
            float sv[4];
            #pragma unroll
            for (int n = 0; n < 4; ++n) {
                float t = S[n][r] * scale;
                sv[n] = (kb + 16 * n + m > q) ? -3.0e30f : t;
            }
            float mx = fmaxf(fmaxf(sv[0], sv[1]), fmaxf(sv[2], sv[3]));
            #pragma unroll
            for (int off = 1; off < 16; off <<= 1)
                mx = fmaxf(mx, __shfl_xor(mx, off, 32));
            const float mnew = fmaxf(mrow[r], mx);
            const float a = __expf(mrow[r] - mnew);
            float sum = 0.0f;
            #pragma unroll
            for (int n = 0; n < 4; ++n) { sv[n] = __expf(sv[n] - mnew); sum += sv[n]; }
            #pragma unroll
            for (int off = 1; off < 16; off <<= 1)
                sum += __shfl_xor(sum, off, 32);
            lrow[r] = lrow[r] * a + sum;
            mrow[r] = mnew;
            alpha[r] = a;
            #pragma unroll
            for (int n = 0; n < 4; ++n)
                myP[(r + 8 * hh) * 64 + 16 * n + m] = (_Float16)sv[n];
        }

        // rescale accumulators while LDS stores drain
        #pragma unroll
        for (int c = 0; c < 8; ++c)
            #pragma unroll
            for (int r = 0; r < 8; ++r) O[c][r] *= alpha[r];

        asm volatile("s_wait_dscnt 0" ::: "memory");

        // P A-frags (two 16x32 key-slices), D-layout -> A-layout via LDS
        v16h pA[2];
        #pragma unroll
        for (int ks = 0; ks < 2; ++ks) {
            const _Float16* prow = myP + m * 64 + 32 * ks;
            v8h lo = *(const v8h*)(prow + hh * 16);
            v8h hi = *(const v8h*)(prow + 8 + hh * 16);
            #pragma unroll
            for (int i = 0; i < 8; ++i) { pA[ks][i] = lo[i]; pA[ks][8 + i] = hi[i]; }
        }

        // ---- O += P @ V : 8 head-dim tiles x 2 key slices, V from LDS ----
        #pragma unroll
        for (int c = 0; c < 8; ++c) {
            #pragma unroll
            for (int ks = 0; ks < 2; ++ks) {
                const _Float16* pv = Vs + (size_t)(16 * c + m) * 64 + 32 * ks + 16 * hh;
                O[c] = wmma16(pA[ks], *(const v16h*)pv, O[c]);
            }
        }

#if USE_TDM
        if (wid == 0) __builtin_amdgcn_s_wait_tensorcnt(0);
#endif
        __syncthreads();
        buf ^= 1;
    }

    // ---- epilogue: normalize and store ----
    float inv[8];
    #pragma unroll
    for (int r = 0; r < 8; ++r) inv[r] = 1.0f / lrow[r];
    float* orow = out + ((size_t)b * SEQ + q0) * HEADSZ;
    #pragma unroll
    for (int c = 0; c < 8; ++c)
        #pragma unroll
        for (int r = 0; r < 8; ++r)
            orow[(size_t)(r + 8 * hh) * HEADSZ + 16 * c + m] = O[c][r] * inv[r];
}

// ---------------------------------------------------------------------------
extern "C" void kernel_launch(void* const* d_in, const int* in_sizes, int n_in,
                              void* d_out, int out_size, void* d_ws, size_t ws_size,
                              hipStream_t stream)
{
    const float* x  = (const float*)d_in[0];
    const float* Wk = (const float*)d_in[1];
    // d_in[2] = W_query: intentionally unused (reference uses key() for q too)
    const float* Wv = (const float*)d_in[3];
    float* out = (float*)d_out;

    _Float16* Kf  = (_Float16*)d_ws;                        // 4 MB
    _Float16* Vt  = Kf  + (size_t)NBATCH * SEQ * HEADSZ;    // 4 MB
    _Float16* WkT = Vt  + (size_t)NBATCH * SEQ * HEADSZ;    // 256 KB
    _Float16* WvT = WkT + (size_t)EMB * HEADSZ;             // 256 KB

    ah_cvt_w_kernel<<<dim3((EMB * HEADSZ) / 256), 256, 0, stream>>>(Wk, Wv, WkT, WvT);
    ah_proj_kernel<<<dim3((NBATCH * SEQ) / 16), 256, 0, stream>>>(x, WkT, WvT, Kf, Vt);
    ah_attn_kernel<<<dim3(NBATCH, SEQ / 64), 128, 0, stream>>>(Kf, Vt, out);
}
